// ForwardTransform_5446018531793
// MI455X (gfx1250) — matrix operand order repaired and checked by reading, not compiled, not run on hardware
//
#include <hip/hip_runtime.h>

// ---------------------------------------------------------------------------
// Sliding-window DFT as split-bf16 WMMA GEMM for gfx1250 (wave32).
//   out[b,f,k] = sum_w d[b,f,w] * (mod_real + i*mod_imag)[w,k]
//   d[b,f,w] = x[b, f*HOP + w - (W-HOP)]  (zero for negative index)
// B fragments double-buffered in LDS via global_load_async_to_lds_b128.
// ---------------------------------------------------------------------------

typedef __attribute__((ext_vector_type(16))) __bf16        v16bf;
typedef __attribute__((ext_vector_type(8)))  float         v8f;
typedef __attribute__((ext_vector_type(4)))  float         v4f;
typedef __attribute__((ext_vector_type(2)))  float         v2f;
typedef __attribute__((ext_vector_type(4)))  unsigned int  v4u;
typedef __attribute__((ext_vector_type(4)))  int           v4i;

#define BB      16
#define HOP     256
#define WLEN    1024
#define NBIN    513
#define TT      524288
#define FF      (TT / HOP)        // 2048 frames
#define MTILES  (BB * FF / 16)    // 2048 M tiles of 16 rows
#define KCHUNKS (WLEN / 32)       // 32 K chunks of 32
#define NTILES  33                // 33 * 16 = 528 >= 513 cols
#define NGROUPS 11                // 3 N-tiles per wave
#define PLANE   ((size_t)KCHUNKS * NTILES * 256)   // u32 per hi/lo plane

union Frag {
    unsigned int u[8];
    v4u          q[2];
    v16bf        v;
};

__device__ __forceinline__ void split_pack(float x0, float x1,
                                           unsigned int& hi, unsigned int& lo) {
    unsigned int u0 = __float_as_uint(x0);
    unsigned int u1 = __float_as_uint(x1);
    float l0f = x0 - __uint_as_float(u0 & 0xffff0000u);
    float l1f = x1 - __uint_as_float(u1 & 0xffff0000u);
    // pack top-16 bits of (x1,x0) -> one dword: v_perm_b32
    hi = __builtin_amdgcn_perm(u1, u0, 0x07060302u);
    lo = __builtin_amdgcn_perm(__float_as_uint(l1f), __float_as_uint(l0f), 0x07060302u);
}

// ---- async global -> LDS (16B per lane), ASYNCcnt-tracked ------------------
__device__ __forceinline__ void async_copy16(const unsigned int* gsrc,
                                             unsigned int* lds_dst) {
#if __has_builtin(__builtin_amdgcn_global_load_async_to_lds_b128)
    typedef __attribute__((address_space(1))) v4i GV;
    typedef __attribute__((address_space(3))) v4i LV;
    __builtin_amdgcn_global_load_async_to_lds_b128(
        (GV*)(unsigned long long)gsrc,
        (LV*)(unsigned int)(unsigned long long)lds_dst,
        0, 0);
#else
    unsigned int loff = (unsigned int)(unsigned long long)lds_dst;
    asm volatile("global_load_async_to_lds_b128 %0, %1, off"
                 :: "v"(loff), "v"(gsrc) : "memory");
#endif
}

__device__ __forceinline__ void wait_async0() {
#if __has_builtin(__builtin_amdgcn_s_wait_asynccnt)
    __builtin_amdgcn_s_wait_asynccnt(0);
#else
    asm volatile("s_wait_asynccnt 0x0" ::: "memory");
#endif
}

// ---------------------------------------------------------------------------
// Prep: pack mod_real/mod_imag into WMMA B-fragment layout, split into
// bf16 hi/lo planes. Layout (u32 units):
//   wsb[ ((q*KCHUNKS + kt)*NTILES + nt)*256 + lane*8 + i ]
//   q: 0=re_hi 1=re_lo 2=im_hi 3=im_lo
//   B fragment in this code's k order (same as A): lane -> n = lane&15, element e -> K =
//   16*(lane>>4) + e.
// ---------------------------------------------------------------------------
__global__ __launch_bounds__(256) void prep_mod_kernel(
        const float* __restrict__ mr, const float* __restrict__ mi,
        unsigned int* __restrict__ wsb) {
    int tid = blockIdx.x * blockDim.x + threadIdx.x;
    const int total = 2 * KCHUNKS * NTILES * 32;
    if (tid >= total) return;
    int lane = tid & 31;
    int t2   = tid >> 5;
    int nt   = t2 % NTILES;
    int t3   = t2 / NTILES;
    int kt   = t3 % KCHUNKS;
    int mat  = t3 / KCHUNKS;                       // 0 = real, 1 = imag
    const float* src = mat ? mi : mr;

    int n  = nt * 16 + (lane & 15);
    int kb = kt * 32 + (lane >> 4) * 16;

    Frag fh, fl;
#pragma unroll
    for (int i = 0; i < 8; ++i) {
        float v0 = 0.0f, v1 = 0.0f;
        if (n < NBIN) {
            v0 = src[(kb + 2 * i + 0) * NBIN + n];
            v1 = src[(kb + 2 * i + 1) * NBIN + n];
        }
        split_pack(v0, v1, fh.u[i], fl.u[i]);
    }
    size_t bhi = ((size_t)(((mat * 2 + 0) * KCHUNKS + kt) * NTILES + nt)) * 256 + lane * 8;
    size_t blo = ((size_t)(((mat * 2 + 1) * KCHUNKS + kt) * NTILES + nt)) * 256 + lane * 8;
    v4u* ph = (v4u*)(wsb + bhi);
    v4u* pl = (v4u*)(wsb + blo);
    ph[0] = fh.q[0]; ph[1] = fh.q[1];
    pl[0] = fl.q[0]; pl[1] = fl.q[1];
}

// ---------------------------------------------------------------------------
// GEMM: one wave -> 16(M) x 48(N) output (3 N-tiles), re+im accumulators.
// 8 waves per block = 8 consecutive M tiles sharing one N-group, so the
// block's 12 B fragments per K-chunk are staged once in LDS (async, 2 bufs).
// grid = (MTILES/8, NGROUPS).
// ---------------------------------------------------------------------------
__global__ __launch_bounds__(256) void stft_wmma_kernel(
        const float* __restrict__ x,
        const unsigned int* __restrict__ wsb,
        float* __restrict__ out) {
    __shared__ unsigned int ldsB[2][12 * 256];     // 2 x 12KB

    const int tid   = threadIdx.x;
    const int lane  = tid & 31;
    const int wave  = tid >> 5;
    const int mtile = blockIdx.x * 8 + wave;
    const int ngrp  = blockIdx.y;
    const int half  = lane >> 4;
    const int mlow  = lane & 15;

    const int b = mtile >> 7;                      // FF/16 = 128 tiles/batch
    const int f = ((mtile & 127) << 4) + mlow;     // this lane's frame (A row)
    const float* xb = x + (size_t)b * TT;

    // ---- per-thread async-copy slots: 3 x 16B covers 12KB per block -------
    size_t       goff[3];
    unsigned int loff[3];
#pragma unroll
    for (int j = 0; j < 3; ++j) {
        int l   = j * 256 + tid;                   // 0..767
        int fid = l >> 6;                          // fragment 0..11 (q*3+nt)
        int q   = fid / 3;
        int nt  = fid % 3;
        int wo  = (l & 63) * 4;                    // u32 offset in fragment
        goff[j] = (size_t)q * PLANE + (size_t)(ngrp * 3 + nt) * 256 + wo;
        loff[j] = (unsigned int)(l * 4);
    }
    auto issue_chunk = [&](int kt, int buf) {
#pragma unroll
        for (int j = 0; j < 3; ++j)
            async_copy16(wsb + goff[j] + (size_t)kt * (NTILES * 256),
                         &ldsB[buf][loff[j]]);
    };

    const v8f z = {0.f, 0.f, 0.f, 0.f, 0.f, 0.f, 0.f, 0.f};
    v8f accR[3], accI[3];
#pragma unroll
    for (int nt = 0; nt < 3; ++nt) { accR[nt] = z; accI[nt] = z; }

    issue_chunk(0, 0);

    for (int kt = 0; kt < KCHUNKS; ++kt) {
        const int cur = kt & 1;
        wait_async0();          // my async writes into ldsB[cur] done
        __syncthreads();        // everyone's done writing cur / reading cur^1
        if (kt + 1 < KCHUNKS) issue_chunk(kt + 1, cur ^ 1);

        // ---- A fragment: 16x32 f32 tile of the delay line, split to bf16 --
        // A k order used here (B matches): lane m = lane&15; element e -> K =
        // 16*(lane>>4) + e. One 16-float run per lane.
        const int kb = kt * 32 + half * 16;
        const int s0 = f * HOP + kb - (WLEN - HOP);   // multiple of 16
        const v4f zf = {0.f, 0.f, 0.f, 0.f};
        v4f r[4];
        if (s0 >= 0) {
            r[0] = *(const v4f*)(xb + s0);
            r[1] = *(const v4f*)(xb + s0 + 4);
        } else { r[0] = zf; r[1] = zf; }
        if (s0 + 8 >= 0) {
            r[2] = *(const v4f*)(xb + s0 + 8);
            r[3] = *(const v4f*)(xb + s0 + 12);
        } else { r[2] = zf; r[3] = zf; }

        Frag ahi, alo;
#pragma unroll
        for (int i = 0; i < 8; ++i) {
            float x0 = r[(2 * i) >> 2][(2 * i) & 3];
            float x1 = r[(2 * i + 1) >> 2][(2 * i + 1) & 3];
            split_pack(x0, x1, ahi.u[i], alo.u[i]);
        }

        // ---- B fragments from LDS + 6 WMMAs per N-tile --------------------
#pragma unroll
        for (int nt = 0; nt < 3; ++nt) {
            const unsigned int* base = &ldsB[cur][0];
            const v4u* pRH = (const v4u*)(base + (0 * 3 + nt) * 256 + lane * 8);
            const v4u* pRL = (const v4u*)(base + (1 * 3 + nt) * 256 + lane * 8);
            const v4u* pIH = (const v4u*)(base + (2 * 3 + nt) * 256 + lane * 8);
            const v4u* pIL = (const v4u*)(base + (3 * 3 + nt) * 256 + lane * 8);
            Frag bRH, bRL, bIH, bIL;
            bRH.q[0] = pRH[0]; bRH.q[1] = pRH[1];
            bRL.q[0] = pRL[0]; bRL.q[1] = pRL[1];
            bIH.q[0] = pIH[0]; bIH.q[1] = pIH[1];
            bIL.q[0] = pIL[0]; bIL.q[1] = pIL[1];

            accR[nt] = __builtin_amdgcn_wmma_f32_16x16x32_bf16(
                false, ahi.v, false, bRH.v, (short)0, accR[nt], false, false);
            accR[nt] = __builtin_amdgcn_wmma_f32_16x16x32_bf16(
                false, alo.v, false, bRH.v, (short)0, accR[nt], false, false);
            accR[nt] = __builtin_amdgcn_wmma_f32_16x16x32_bf16(
                false, ahi.v, false, bRL.v, (short)0, accR[nt], false, false);

            accI[nt] = __builtin_amdgcn_wmma_f32_16x16x32_bf16(
                false, ahi.v, false, bIH.v, (short)0, accI[nt], false, false);
            accI[nt] = __builtin_amdgcn_wmma_f32_16x16x32_bf16(
                false, alo.v, false, bIH.v, (short)0, accI[nt], false, false);
            accI[nt] = __builtin_amdgcn_wmma_f32_16x16x32_bf16(
                false, ahi.v, false, bIL.v, (short)0, accI[nt], false, false);
        }
    }

    // ---- epilogue: C/D layout lane n = lane&15, VGPR v -> m = v + 8*half.
    // complex64 interleave: (re,im) pair as one 64-bit store.
#pragma unroll
    for (int nt = 0; nt < 3; ++nt) {
        const int n = (ngrp * 3 + nt) * 16 + mlow;
        if (n < NBIN) {
#pragma unroll
            for (int v = 0; v < 8; ++v) {
                const int mg = mtile * 16 + half * 8 + v;   // = b*FF + frame
                v2f val = {accR[nt][v], accI[nt][v]};
                *(v2f*)(out + ((size_t)mg * NBIN + n) * 2) = val;
            }
        }
    }
}

extern "C" void kernel_launch(void* const* d_in, const int* in_sizes, int n_in,
                              void* d_out, int out_size, void* d_ws, size_t ws_size,
                              hipStream_t stream) {
    const float* x  = (const float*)d_in[0];
    const float* mr = (const float*)d_in[1];
    const float* mi = (const float*)d_in[2];
    float* out = (float*)d_out;
    unsigned int* wsb = (unsigned int*)d_ws;   // 4 planes * 1.08 MB = 4.3 MB

    const int prep_threads = 2 * KCHUNKS * NTILES * 32;   // 67584
    prep_mod_kernel<<<(prep_threads + 255) / 256, 256, 0, stream>>>(mr, mi, wsb);

    dim3 grid(MTILES / 8, NGROUPS);
    stft_wmma_kernel<<<grid, 256, 0, stream>>>(x, wsb, out);
}
